// SO2_Convolution_70824010711660
// MI455X (gfx1250) — matrix operand order repaired and checked by reading, not compiled, not run on hardware
//
#include <hip/hip_runtime.h>

typedef float  v2f  __attribute__((ext_vector_type(2)));
typedef float  v8f  __attribute__((ext_vector_type(8)));
typedef __bf16 v8bf __attribute__((ext_vector_type(8)));
typedef __bf16 v16bf __attribute__((ext_vector_type(16)));
typedef unsigned int   uint32;
typedef unsigned short ushort16;

#define NROWS 29
#define CCH 64
#define ESTRIDE (NROWS * CCH) /* 1856 floats per edge */

// Weight element counts
#define NW0 (448 * 448)
#define NW1 (768 * 384)
#define NW2 (640 * 320)

// mp-row r -> lp-row perm[r]; same permutation serves the input gather and the
// output scatter (FROM_M = argsort(TO_M) => out mp-row r lands at lp-row TO_M[r]).
__constant__ int c_perm[NROWS] = {
    0, 2, 6, 11, 16, 21, 26,       // m=0  rows 0..6
    3, 7, 12, 17, 22, 27,          // m=+1 rows 7..12
    1, 5, 10, 15, 20, 25,          // m=-1 rows 13..18
    8, 13, 18, 23, 28,             // m=+2 rows 19..23
    4, 9, 14, 19, 24               // m=-2 rows 24..28
};

// ---------------- helpers ----------------
static __device__ __forceinline__ ushort16 bf16_rne(float f) {
    uint32 u = __builtin_bit_cast(uint32, f);
    u += 0x7fffu + ((u >> 16) & 1u);
    return (ushort16)(u >> 16);
}
static __device__ __forceinline__ float bf16_up(ushort16 h) {
    return __builtin_bit_cast(float, (uint32)h << 16);
}
static __device__ __forceinline__ v8f wmma_bf16(v16bf a, v16bf b, v8f c) {
    return __builtin_amdgcn_wmma_f32_16x16x32_bf16(false, a, false, b, (short)0, c,
                                                   false, false);
}
static __device__ __forceinline__ v8f wmma_f32(v2f a, v2f b, v8f c) {
    return __builtin_amdgcn_wmma_f32_16x16x4_f32(false, a, false, b, (short)0, c,
                                                 false, false);
}
// A-fragment (16x32 bf16): lane holds K [base..base+15] contiguous
static __device__ __forceinline__ v16bf load_afrag(const ushort16* p) {
    v8bf a = *reinterpret_cast<const v8bf*>(p);
    v8bf b = *reinterpret_cast<const v8bf*>(p + 8);
    return __builtin_shufflevector(a, b, 0, 1, 2, 3, 4, 5, 6, 7,
                                   8, 9, 10, 11, 12, 13, 14, 15);
}

// ---------------- weight split: W -> bf16 hi/lo planes ----------------
__global__ void split_weights_kernel(const float* __restrict__ W,
                                     ushort16* __restrict__ hi,
                                     ushort16* __restrict__ lo, int n) {
    int i = blockIdx.x * blockDim.x + threadIdx.x;
    if (i < n) {
        float w = W[i];
        ushort16 h = bf16_rne(w);
        hi[i] = h;
        lo[i] = bf16_rne(w - bf16_up(h));
    }
}

// ---------------- main bf16-split kernel ----------------
// grid: x = 18 column groups (7 for m=0, 6 for m=1, 5 for m=2)  [fast dim -> L2 reuse]
//       y = edge tiles of 16
__global__ __launch_bounds__(128, 1) void so2_conv_bf16_kernel(
    const float* __restrict__ x,   // (E, 29, 64)
    const ushort16* __restrict__ W0h, const ushort16* __restrict__ W0l,
    const ushort16* __restrict__ W1h, const ushort16* __restrict__ W1l,
    const ushort16* __restrict__ W2h, const ushort16* __restrict__ W2l,
    const float* __restrict__ b0,  // (448)
    float* __restrict__ out,       // (E, 29, 64)
    int E) {
    // hi/lo bf16 planes, stride = K2+8 (dword stride == 4 or 36 mod 64 -> conflict-free)
    __shared__ ushort16 AsH[16 * 776];
    __shared__ ushort16 AsL[16 * 776];

    const int tid  = threadIdx.x;
    const int wave = tid >> 5;
    const int lane = tid & 31;
    const int e0   = blockIdx.y * 16;

    int seg, cg;
    if (blockIdx.x < 7)       { seg = 0; cg = blockIdx.x; }
    else if (blockIdx.x < 13) { seg = 1; cg = blockIdx.x - 7; }
    else                      { seg = 2; cg = blockIdx.x - 13; }

    int K, rowbase, nrows, half;
    const ushort16 *Wh, *Wl;
    if (seg == 0)      { K = 448; rowbase = 0;  nrows = 7;  Wh = W0h; Wl = W0l; half = 0;   }
    else if (seg == 1) { K = 384; rowbase = 7;  nrows = 12; Wh = W1h; Wl = W1l; half = 384; }
    else               { K = 320; rowbase = 19; nrows = 10; Wh = W2h; Wl = W2l; half = 320; }

    const int K2     = nrows * CCH;
    const int stride = K2 + 8;

    // ---- cooperative gather + fp32 -> bf16 hi/lo split into LDS ----
    const int q_per_edge = K2 >> 2;
    for (int i = tid; i < 16 * q_per_edge; i += 128) {
        const int edge = i / q_per_edge;
        const int kf   = (i - edge * q_per_edge) << 2;
        const int r    = kf >> 6;
        const int ch   = kf & 63;
        int e = e0 + edge;
        if (e >= E) e = E - 1;
        const float4 v = *reinterpret_cast<const float4*>(
            x + (size_t)e * ESTRIDE + (size_t)c_perm[rowbase + r] * CCH + ch);
        ushort16 h0 = bf16_rne(v.x), h1 = bf16_rne(v.y);
        ushort16 h2 = bf16_rne(v.z), h3 = bf16_rne(v.w);
        uint2 ph, pl;
        ph.x = (uint32)h0 | ((uint32)h1 << 16);
        ph.y = (uint32)h2 | ((uint32)h3 << 16);
        pl.x = (uint32)bf16_rne(v.x - bf16_up(h0)) |
               ((uint32)bf16_rne(v.y - bf16_up(h1)) << 16);
        pl.y = (uint32)bf16_rne(v.z - bf16_up(h2)) |
               ((uint32)bf16_rne(v.w - bf16_up(h3)) << 16);
        *reinterpret_cast<uint2*>(&AsH[edge * stride + kf]) = ph;
        *reinterpret_cast<uint2*>(&AsL[edge * stride + kf]) = pl;
    }
    __syncthreads();

    const int m16   = lane & 15;
    const int khi8  = (lane >> 4) << 4;   // A-fragment K sub-offset (0 or 16)
    const int khi16 = (lane >> 4) << 4;   // B-fragment K sub-offset (0 or 16)
    const int col0  = cg * 64 + wave * 16;
    const int erow  = 8 * (lane >> 4);    // C/D: vgpr v holds edge row v+erow

    if (seg == 0) {
        v8f acc = {0.f, 0.f, 0.f, 0.f, 0.f, 0.f, 0.f, 0.f};
        const ushort16* Wph = Wh + (size_t)(col0 + m16) * 448 + khi16;
        const ushort16* Wpl = Wl + (size_t)(col0 + m16) * 448 + khi16;
        const ushort16* Ah  = &AsH[m16 * stride + khi8];
        const ushort16* Al  = &AsL[m16 * stride + khi8];
#pragma unroll 2
        for (int k = 0; k < 448; k += 32) {
            v16bf ah = load_afrag(Ah + k);
            v16bf al = load_afrag(Al + k);
            v16bf bh = *reinterpret_cast<const v16bf*>(Wph + k);
            v16bf bl = *reinterpret_cast<const v16bf*>(Wpl + k);
            acc = wmma_bf16(ah, bh, acc);   // hi*hi
            acc = wmma_bf16(ah, bl, acc);   // hi*lo
            acc = wmma_bf16(al, bh, acc);   // lo*hi
        }
        const float bias = b0[col0 + m16];
        const int lp  = c_perm[col0 >> 6];
        const int och = (col0 & 63) + m16;
        float* ob = out + (size_t)(e0 + erow) * ESTRIDE + lp * CCH + och;
#pragma unroll
        for (int v = 0; v < 8; ++v)
            if (e0 + erow + v < E) ob[(size_t)v * ESTRIDE] = acc[v] + bias;
    } else {
        v8f aR0 = {0.f,0.f,0.f,0.f,0.f,0.f,0.f,0.f};
        v8f aR1 = aR0, aI0 = aR0, aI1 = aR0;
        const ushort16* WRh = Wh + (size_t)(col0 + m16) * K + khi16;
        const ushort16* WRl = Wl + (size_t)(col0 + m16) * K + khi16;
        const ushort16* WIh = Wh + (size_t)(col0 + half + m16) * K + khi16;
        const ushort16* WIl = Wl + (size_t)(col0 + half + m16) * K + khi16;
        const ushort16* Ah0 = &AsH[m16 * stride + khi8];
        const ushort16* Al0 = &AsL[m16 * stride + khi8];
        for (int k = 0; k < K; k += 32) {
            v16bf ah0 = load_afrag(Ah0 + k);
            v16bf al0 = load_afrag(Al0 + k);
            v16bf ah1 = load_afrag(Ah0 + K + k);     // n=1 block is K bf16 later
            v16bf al1 = load_afrag(Al0 + K + k);
            v16bf bRh = *reinterpret_cast<const v16bf*>(WRh + k);
            v16bf bRl = *reinterpret_cast<const v16bf*>(WRl + k);
            v16bf bIh = *reinterpret_cast<const v16bf*>(WIh + k);
            v16bf bIl = *reinterpret_cast<const v16bf*>(WIl + k);
            aR0 = wmma_bf16(ah0, bRh, aR0);
            aR0 = wmma_bf16(ah0, bRl, aR0);
            aR0 = wmma_bf16(al0, bRh, aR0);
            aR1 = wmma_bf16(ah1, bRh, aR1);
            aR1 = wmma_bf16(ah1, bRl, aR1);
            aR1 = wmma_bf16(al1, bRh, aR1);
            aI0 = wmma_bf16(ah0, bIh, aI0);
            aI0 = wmma_bf16(ah0, bIl, aI0);
            aI0 = wmma_bf16(al0, bIh, aI0);
            aI1 = wmma_bf16(ah1, bIh, aI1);
            aI1 = wmma_bf16(ah1, bIl, aI1);
            aI1 = wmma_bf16(al1, bIh, aI1);
        }
        const int sz   = nrows >> 1;
        const int lp_r = c_perm[rowbase + (col0 >> 6)];
        const int lp_i = c_perm[rowbase + sz + (col0 >> 6)];
        const int och  = (col0 & 63) + m16;
        float* pr = out + (size_t)(e0 + erow) * ESTRIDE + lp_r * CCH + och;
        float* pi = out + (size_t)(e0 + erow) * ESTRIDE + lp_i * CCH + och;
#pragma unroll
        for (int v = 0; v < 8; ++v) {
            if (e0 + erow + v < E) {
                pr[(size_t)v * ESTRIDE] = aR0[v] - aI1[v];  // x_m_r
                pi[(size_t)v * ESTRIDE] = aR1[v] + aI0[v];  // x_m_i
            }
        }
    }
}

// ---------------- exact-fp32 fallback (used if d_ws too small) ----------------
__global__ __launch_bounds__(128, 1) void so2_conv_f32_kernel(
    const float* __restrict__ x, const float* __restrict__ W0,
    const float* __restrict__ b0, const float* __restrict__ W1,
    const float* __restrict__ W2, float* __restrict__ out, int E) {
    __shared__ float As[16 * 772];
    const int tid = threadIdx.x, wave = tid >> 5, lane = tid & 31;
    const int e0 = blockIdx.y * 16;
    int seg, cg;
    if (blockIdx.x < 7)       { seg = 0; cg = blockIdx.x; }
    else if (blockIdx.x < 13) { seg = 1; cg = blockIdx.x - 7; }
    else                      { seg = 2; cg = blockIdx.x - 13; }
    int K, rowbase, nrows, half;
    const float* W;
    if (seg == 0)      { K = 448; rowbase = 0;  nrows = 7;  W = W0; half = 0;   }
    else if (seg == 1) { K = 384; rowbase = 7;  nrows = 12; W = W1; half = 384; }
    else               { K = 320; rowbase = 19; nrows = 10; W = W2; half = 320; }
    const int K2 = nrows * CCH, stride = K2 + 4;
    const int q_per_edge = K2 >> 2;
    for (int i = tid; i < 16 * q_per_edge; i += 128) {
        const int edge = i / q_per_edge;
        const int kf = (i - edge * q_per_edge) << 2;
        int e = e0 + edge; if (e >= E) e = E - 1;
        const float4 v = *reinterpret_cast<const float4*>(
            x + (size_t)e * ESTRIDE + (size_t)c_perm[rowbase + (kf >> 6)] * CCH + (kf & 63));
        *reinterpret_cast<float4*>(&As[edge * stride + kf]) = v;
    }
    __syncthreads();
    const int m16 = lane & 15, khi = (lane >> 4) << 1;
    const int col0 = cg * 64 + wave * 16, erow = 8 * (lane >> 4);
    if (seg == 0) {
        v8f acc = {0.f,0.f,0.f,0.f,0.f,0.f,0.f,0.f};
        const float* Wrow = W + (size_t)(col0 + m16) * 448 + khi;
        const float* Ab = &As[m16 * stride + khi];
#pragma unroll 8
        for (int k = 0; k < 448; k += 4)
            acc = wmma_f32(*(const v2f*)(Ab + k), *(const v2f*)(Wrow + k), acc);
        const float bias = b0[col0 + m16];
        float* ob = out + (size_t)(e0 + erow) * ESTRIDE + c_perm[col0 >> 6] * CCH +
                    (col0 & 63) + m16;
#pragma unroll
        for (int v = 0; v < 8; ++v)
            if (e0 + erow + v < E) ob[(size_t)v * ESTRIDE] = acc[v] + bias;
    } else {
        v8f aR0 = {0.f,0.f,0.f,0.f,0.f,0.f,0.f,0.f};
        v8f aR1 = aR0, aI0 = aR0, aI1 = aR0;
        const float* WR = W + (size_t)(col0 + m16) * K + khi;
        const float* WI = W + (size_t)(col0 + half + m16) * K + khi;
        const float* A0 = &As[m16 * stride + khi];
        const float* A1 = A0 + K;
#pragma unroll 4
        for (int k = 0; k < K; k += 4) {
            v2f br = *(const v2f*)(WR + k), bi = *(const v2f*)(WI + k);
            v2f a0 = *(const v2f*)(A0 + k), a1 = *(const v2f*)(A1 + k);
            aR0 = wmma_f32(a0, br, aR0); aR1 = wmma_f32(a1, br, aR1);
            aI0 = wmma_f32(a0, bi, aI0); aI1 = wmma_f32(a1, bi, aI1);
        }
        const int sz = nrows >> 1, och = (col0 & 63) + m16;
        float* pr = out + (size_t)(e0 + erow) * ESTRIDE +
                    c_perm[rowbase + (col0 >> 6)] * CCH + och;
        float* pi = out + (size_t)(e0 + erow) * ESTRIDE +
                    c_perm[rowbase + sz + (col0 >> 6)] * CCH + och;
#pragma unroll
        for (int v = 0; v < 8; ++v)
            if (e0 + erow + v < E) {
                pr[(size_t)v * ESTRIDE] = aR0[v] - aI1[v];
                pi[(size_t)v * ESTRIDE] = aR1[v] + aI0[v];
            }
    }
}

extern "C" void kernel_launch(void* const* d_in, const int* in_sizes, int n_in,
                              void* d_out, int out_size, void* d_ws, size_t ws_size,
                              hipStream_t stream) {
    const float* x  = (const float*)d_in[0];
    // d_in[1] = x_edge : unused by the reference computation
    const float* W0 = (const float*)d_in[2];
    const float* b0 = (const float*)d_in[3];
    const float* W1 = (const float*)d_in[4];
    const float* W2 = (const float*)d_in[5];
    float* out = (float*)d_out;

    const int E = in_sizes[0] / ESTRIDE;  // 50000
    dim3 grid(18, (E + 15) / 16);         // x: column groups (fast) -> L2 reuse of edge tile

    const size_t need = (size_t)2 * (NW0 + NW1 + NW2) * sizeof(ushort16);  // ~2.8 MB
    if (ws_size >= need) {
        ushort16* w0h = (ushort16*)d_ws;
        ushort16* w0l = w0h + NW0;
        ushort16* w1h = w0l + NW0;
        ushort16* w1l = w1h + NW1;
        ushort16* w2h = w1l + NW1;
        ushort16* w2l = w2h + NW2;
        split_weights_kernel<<<(NW0 + 255) / 256, 256, 0, stream>>>(W0, w0h, w0l, NW0);
        split_weights_kernel<<<(NW1 + 255) / 256, 256, 0, stream>>>(W1, w1h, w1l, NW1);
        split_weights_kernel<<<(NW2 + 255) / 256, 256, 0, stream>>>(W2, w2h, w2l, NW2);
        so2_conv_bf16_kernel<<<grid, 128, 0, stream>>>(x, w0h, w0l, w1h, w1l, w2h, w2l,
                                                       b0, out, E);
    } else {
        so2_conv_f32_kernel<<<grid, 128, 0, stream>>>(x, W0, b0, W1, W2, out, E);
    }
}
